// word_kv_memory_3289944949026
// MI455X (gfx1250) — matrix operand order repaired and checked by reading, not compiled, not run on hardware
//
#include <hip/hip_runtime.h>
#include <hip/hip_bf16.h>
#include <stdint.h>

// ---------- types ----------
typedef __attribute__((ext_vector_type(16))) __bf16 v16bf;
typedef __attribute__((ext_vector_type(8)))  __bf16 v8bf;
typedef __attribute__((ext_vector_type(8)))  float  v8f;
typedef unsigned int uint4v __attribute__((ext_vector_type(4)));
typedef int          int8v  __attribute__((ext_vector_type(8)));
typedef int          int4v  __attribute__((ext_vector_type(4)));

#define B_   32
#define LQ_  256
#define LKV_ 256
#define HID_ 1024
#define EMB_ 1024

// ---------- scalar helpers ----------
static __device__ __forceinline__ __bf16 f2bf(float f) {
    unsigned int x = __float_as_uint(f);
    unsigned int r = (x + 0x7FFFu + ((x >> 16) & 1u)) >> 16;   // RNE
    union { unsigned short s; __bf16 b; } u; u.s = (unsigned short)r;
    return u.b;
}
static __device__ __forceinline__ float bf2f(__bf16 b) {
    union { __bf16 b; unsigned short s; } u; u.b = b;
    return __uint_as_float(((unsigned int)u.s) << 16);
}
static __device__ __forceinline__ v8f vzero() {
    v8f z;
#pragma unroll
    for (int i = 0; i < 8; i++) z[i] = 0.0f;
    return z;
}
// A-operand fragment (16x32 bf16): lane holds A[row][kb..kb+15] (16 contiguous)
static __device__ __forceinline__ v16bf load_a16(const __bf16* rowptr, int kb) {
    v8bf lo = *(const v8bf*)(rowptr + kb);
    v8bf hi = *(const v8bf*)(rowptr + kb + 8);
    v16bf r;
#pragma unroll
    for (int i = 0; i < 8; i++) { r[i] = lo[i]; r[i + 8] = hi[i]; }
    return r;
}
// B-operand fragment (32x16 bf16): lane holds Bt[n][kb2 .. kb2+15] (16 contiguous)
static __device__ __forceinline__ v16bf load_b16(const __bf16* p) {
    v8bf lo = *(const v8bf*)(p);
    v8bf hi = *(const v8bf*)(p + 8);
    v16bf r;
#pragma unroll
    for (int i = 0; i < 8; i++) { r[i] = lo[i]; r[i + 8] = hi[i]; }
    return r;
}
#define WMMA_BF16(A, Bm, C) \
    __builtin_amdgcn_wmma_f32_16x16x32_bf16(false, (A), false, (Bm), (short)0, (C), false, false)

// ---------- CDNA5 async copy to LDS (ASYNCcnt) ----------
static __device__ __forceinline__ unsigned ldsoff(const void* p) {
    return (unsigned)(uintptr_t)p;     // low 32 bits of generic ptr == LDS byte address
}
static __device__ __forceinline__ void async_b128(unsigned lds_addr, const void* gptr) {
    asm volatile("global_load_async_to_lds_b128 %0, %1, off"
                 :: "v"(lds_addr), "v"((unsigned long long)(uintptr_t)gptr)
                 : "memory");
}
static __device__ __forceinline__ void wait_async0() {
    asm volatile("s_wait_asynccnt 0x0" ::: "memory");
}

// ---------- CDNA5 TDM 2D tile load (TENSORcnt), ISA section 8 D# layout ----------
static __device__ __forceinline__ void tdm_load_2d(
    unsigned lds_addr, const void* gbase,
    unsigned tile_x, unsigned tile_y,            // tile dims (elements)
    unsigned tensor_x, unsigned tensor_y,        // tensor dims (elements)
    unsigned stride_x)                           // dim0 stride (elements)
{
    unsigned long long ga = (unsigned long long)(uintptr_t)gbase;
    uint4v g0;
    g0[0] = 1u;                                             // count=1, user-mode
    g0[1] = lds_addr;                                       // lds_addr (bytes)
    g0[2] = (unsigned)(ga & 0xFFFFFFFFu);                   // global_addr[31:0]
    g0[3] = (unsigned)((ga >> 32) & 0x01FFFFFFu) | (2u << 30); // [56:32] | type=2
    int8v g1;
    g1[0] = (int)(1u << 16);                                // data_size=1 (2 bytes)
    g1[1] = (int)((tensor_x & 0xFFFFu) << 16);              // tensor_dim0[15:0]
    g1[2] = (int)((tensor_x >> 16) | ((tensor_y & 0xFFFFu) << 16));
    g1[3] = (int)((tensor_y >> 16) | (tile_x << 16));       // tile_dim0
    g1[4] = (int)(tile_y & 0xFFFFu);                        // tile_dim1 (tile_dim2=0)
    g1[5] = (int)stride_x;                                  // tensor_dim0_stride[31:0]
    g1[6] = 0;
    g1[7] = 0;
    int4v z4 = {0, 0, 0, 0};
#if defined(__clang_major__) && (__clang_major__ >= 23)
    int8v z8 = {0, 0, 0, 0, 0, 0, 0, 0};
    __builtin_amdgcn_tensor_load_to_lds(g0, g1, z4, z4, z8, 0);
#else
    __builtin_amdgcn_tensor_load_to_lds(g0, g1, z4, z4, 0);
#endif
}

// =====================================================================
// Kernel 0a: f32 -> bf16 bulk convert (hidden, fc_w).  One-time pass.
// =====================================================================
__global__ __launch_bounds__(256) void k0_cvt(const float* __restrict__ src,
                                              __bf16* __restrict__ dst, int n)
{
    int i = (blockIdx.x * 256 + threadIdx.x) * 4;
    if (i + 3 < n) {
        float4 v = *(const float4*)(src + i);
        dst[i + 0] = f2bf(v.x); dst[i + 1] = f2bf(v.y);
        dst[i + 2] = f2bf(v.z); dst[i + 3] = f2bf(v.w);
    }
}

// =====================================================================
// Kernel 0b: gather embedding rows to bf16.
//   eaBuf[b*256+k][h] = bf16(Wa[captions[b,k]][h])   (row-major)
//   ecT [b][h][k]     = bf16(Wc[tfeat[b,k]][h])      (pre-transposed)
// =====================================================================
__global__ __launch_bounds__(256) void k0_gather(
    const int* __restrict__ cap, const int* __restrict__ tf,
    const float* __restrict__ Wa, const float* __restrict__ Wc,
    __bf16* __restrict__ eaBuf, __bf16* __restrict__ ecT)
{
    int bk = blockIdx.x;               // b*256 + k
    int b = bk >> 8, k = bk & 255;
    int h0 = threadIdx.x * 4;
    float4 va = *(const float4*)(Wa + (size_t)cap[bk] * HID_ + h0);
    float4 vc = *(const float4*)(Wc + (size_t)tf[bk] * HID_ + h0);
    __bf16* ea = eaBuf + (size_t)bk * HID_ + h0;
    ea[0] = f2bf(va.x); ea[1] = f2bf(va.y); ea[2] = f2bf(va.z); ea[3] = f2bf(va.w);
    size_t e = ((size_t)b * HID_ + h0) * 256 + k;
    ecT[e +   0] = f2bf(vc.x);
    ecT[e + 256] = f2bf(vc.y);
    ecT[e + 512] = f2bf(vc.z);
    ecT[e + 768] = f2bf(vc.w);
}

// =====================================================================
// Kernel 1: attention.  One WG per (batch b, 64-row q-tile).
// Double-buffered async LDS staging feeding bf16 WMMA.
// NOTE: no local arrays of LDS-derived pointers (ld.lld cannot relocate
// addrspacecast constants) -- buffer bases are computed arithmetically.
// =====================================================================
__global__ __launch_bounds__(256) void k1_attn(
    const __bf16* __restrict__ hidBf, const __bf16* __restrict__ eaBuf,
    const __bf16* __restrict__ ecT, const float* __restrict__ fmask,
    __bf16* __restrict__ oOut)
{
    const int tid  = threadIdx.x;
    const int lane = tid & 31;
    const int w    = tid >> 5;
    const int b    = blockIdx.x >> 2;
    const int qt   = blockIdx.x & 3;
    const int qbase = qt * 64;

    __shared__ float sInv[64];
    __shared__ __align__(16) __bf16 sDelta[64 * 256];    // 32 KB
    __shared__ __align__(16) __bf16 sStage[20480];       // 40 KB (double buffers)
    // phase-1 layout: sA(bu) = sStage + bu*2048      (64x32 each)
    //                 sB(bu) = sStage + 4096 + bu*8192 (256x32 each)
    // phase-2 layout: sE(bu) = sStage + bu*8448      ([32][264] padded)

    const int arow  = lane & 15;
    const int akb   = (lane < 16) ? 0 : 16;
    const int bkb   = (lane < 16) ? 0 : 16;
    const int mrow0 = (lane < 16) ? 0 : 8;
    const int ncol  = lane & 15;

    const int sti  = tid >> 2;         // row 0..63
    const int stc  = (tid & 3) * 8;    // col chunk of 8

    // ---- phase 1: u = H @ ea^T ----
    const int rt     = w >> 1;
    const int ctbase = (w & 1) * 8;
    v8f acc[8];
#pragma unroll
    for (int t = 0; t < 8; t++) acc[t] = vzero();

    // prologue: stage step 0 into buffer 0
    {
        async_b128(ldsoff(sStage + sti * 32 + stc),
                   hidBf + (size_t)(b * LQ_ + qbase + sti) * HID_ + 0 + stc);
        for (int rep = 0; rep < 4; rep++) {
            int idx = rep * 256 + tid, k = idx >> 2, jc = (idx & 3) * 8;
            async_b128(ldsoff(sStage + 4096 + k * 32 + jc),
                       eaBuf + (size_t)(b * LKV_ + k) * HID_ + 0 + jc);
        }
        wait_async0();
        __syncthreads();
    }
    for (int s = 0; s < HID_ / 32; s++) {
        const int bu = s & 1;
        __bf16* sAc = sStage + bu * 2048;
        __bf16* sBc = sStage + 4096 + bu * 8192;
        if (s + 1 < HID_ / 32) {               // prefetch next chunk
            int h0 = (s + 1) * 32;
            __bf16* sAn = sStage + (bu ^ 1) * 2048;
            __bf16* sBn = sStage + 4096 + (bu ^ 1) * 8192;
            async_b128(ldsoff(sAn + sti * 32 + stc),
                       hidBf + (size_t)(b * LQ_ + qbase + sti) * HID_ + h0 + stc);
            for (int rep = 0; rep < 4; rep++) {
                int idx = rep * 256 + tid, k = idx >> 2, jc = (idx & 3) * 8;
                async_b128(ldsoff(sBn + k * 32 + jc),
                           eaBuf + (size_t)(b * LKV_ + k) * HID_ + h0 + jc);
            }
        }
        v16bf af = load_a16(sAc + (rt * 16 + arow) * 32, akb);
#pragma unroll
        for (int t = 0; t < 8; t++) {
            v16bf bfv = load_b16(sBc + ((ctbase + t) * 16 + arow) * 32 + bkb);
            acc[t] = WMMA_BF16(af, bfv, acc[t]);
        }
        wait_async0();
        __syncthreads();
    }

    // ---- delta = exp(u)*clip(mask,0,1) -> bf16 LDS ----
#pragma unroll
    for (int t = 0; t < 8; t++) {
        int kloc = (ctbase + t) * 16 + ncol;
#pragma unroll
        for (int r = 0; r < 8; r++) {
            int qloc = rt * 16 + mrow0 + r;
            float m = fmask[(((size_t)b * LQ_) + qbase + qloc) * LKV_ + kloc];
            m = fminf(fmaxf(m, 0.0f), 1.0f);
            sDelta[qloc * 256 + kloc] = f2bf(__expf(acc[t][r]) * m);
        }
    }
    __syncthreads();
    if (tid < 64) {
        float s = 0.0f;
        for (int k = 0; k < 256; k++) s += bf2f(sDelta[tid * 256 + k]);
        sInv[tid] = 1.0f / (s + 1e-10f);
    }
    __syncthreads();

    // ---- phase 2: o = delta @ ec, 32-wide h-chunks, ecT pre-transposed ----
    const int rt2 = w >> 1;
    const int ct2 = w & 1;
    {
        for (int rep = 0; rep < 4; rep++) {      // 32 rows x 256 cols, 4 x b128/thread
            int idx = rep * 256 + tid, jh = idx >> 5, pk = (idx & 31) * 8;
            async_b128(ldsoff(sStage + jh * 264 + pk),
                       ecT + ((size_t)(b * HID_ + 0 + jh)) * 256 + pk);
        }
        wait_async0();
        __syncthreads();
    }
    for (int hc = 0; hc < HID_ / 32; hc++) {
        const int bu = hc & 1;
        __bf16* sEc = sStage + bu * 8448;
        if (hc + 1 < HID_ / 32) {
            int hb = (hc + 1) * 32;
            __bf16* sEn = sStage + (bu ^ 1) * 8448;
            for (int rep = 0; rep < 4; rep++) {
                int idx = rep * 256 + tid, jh = idx >> 5, pk = (idx & 31) * 8;
                async_b128(ldsoff(sEn + jh * 264 + pk),
                           ecT + ((size_t)(b * HID_ + hb + jh)) * 256 + pk);
            }
        }
        v8f oacc = vzero();
#pragma unroll
        for (int kv0 = 0; kv0 < 256; kv0 += 32) {
            v16bf af  = load_a16(sDelta + (rt2 * 16 + arow) * 256 + kv0, akb);
            v16bf bfv = load_b16(sEc + (ct2 * 16 + arow) * 264 + kv0 + bkb);
            oacc = WMMA_BF16(af, bfv, oacc);
        }
        int hb = hc * 32;
#pragma unroll
        for (int r = 0; r < 8; r++) {
            int qloc = rt2 * 16 + mrow0 + r;
            float val = oacc[r] * sInv[qloc];
            oOut[((size_t)(b * LQ_ + qbase + qloc)) * HID_ + hb + ct2 * 16 + ncol] = f2bf(val);
        }
        wait_async0();
        __syncthreads();
    }
}

// =====================================================================
// Kernel 2: FC out[8192,1024] = concat(hid,o)@fc_w^T + b + 1e-10 (pre-norm).
// X staged via per-lane async copies; fc_w tiles via TDM tensor_load_to_lds.
// =====================================================================
__global__ __launch_bounds__(256) void k2_fc(
    const __bf16* __restrict__ hidBf, const __bf16* __restrict__ oIn,
    const __bf16* __restrict__ fcwBf, const float* __restrict__ fc_b,
    float* __restrict__ out)
{
    const int tid  = threadIdx.x;
    const int lane = tid & 31;
    const int w    = tid >> 5;
    const int r0   = blockIdx.x * 64;

    __shared__ __align__(16) __bf16 sX[2][64 * 32];     // 4 KB x2
    __shared__ __align__(16) __bf16 sW[2][128 * 32];    // 8 KB x2

    const int arow  = lane & 15;
    const int akb   = (lane < 16) ? 0 : 16;
    const int bkb   = (lane < 16) ? 0 : 16;
    const int mrow0 = (lane < 16) ? 0 : 8;
    const int rt     = w & 3;
    const int cthalf = w >> 2;

    const int sti = tid >> 2;          // X row 0..63
    const int stc = (tid & 3) * 8;     // X col chunk of 8

    for (int chunk = 0; chunk < 8; chunk++) {
        v8f acc[4];
#pragma unroll
        for (int t = 0; t < 4; t++) acc[t] = vzero();

        // prologue: stage ks=0 into buffer 0
        {
            const __bf16* src = hidBf + (size_t)(r0 + sti) * HID_ + stc;   // c0 = 0
            async_b128(ldsoff(&sX[0][sti * 32 + stc]), src);
            if (w == 0)
                tdm_load_2d(ldsoff(&sW[0][0]),
                            fcwBf + (size_t)(chunk * 128) * (2 * HID_) + 0,
                            /*tile*/ 32, 128, /*tensor*/ 2 * HID_, EMB_,
                            /*stride*/ 2 * HID_);
            wait_async0();
            if (w == 0) __builtin_amdgcn_s_wait_tensorcnt(0);
            __syncthreads();
        }
        for (int ks = 0; ks < 64; ks++) {
            const int bu = ks & 1;
            if (ks + 1 < 64) {
                int c0 = (ks + 1) * 32;
                const __bf16* src = (c0 < HID_)
                    ? hidBf + (size_t)(r0 + sti) * HID_ + c0 + stc
                    : oIn   + (size_t)(r0 + sti) * HID_ + (c0 - HID_) + stc;
                async_b128(ldsoff(&sX[bu ^ 1][sti * 32 + stc]), src);
                if (w == 0)
                    tdm_load_2d(ldsoff(&sW[bu ^ 1][0]),
                                fcwBf + (size_t)(chunk * 128) * (2 * HID_) + c0,
                                32, 128, 2 * HID_, EMB_, 2 * HID_);
            }
            v16bf af = load_a16(&sX[bu][(rt * 16 + arow) * 32], akb);
#pragma unroll
            for (int t = 0; t < 4; t++) {
                int eloc = (cthalf * 4 + t) * 16 + arow;
                v16bf bfv = load_b16(&sW[bu][eloc * 32 + bkb]);
                acc[t] = WMMA_BF16(af, bfv, acc[t]);
            }
            wait_async0();
            if (w == 0) __builtin_amdgcn_s_wait_tensorcnt(0);
            __syncthreads();
        }
        // epilogue: + bias + 1e-10 -> pre-norm f32
#pragma unroll
        for (int t = 0; t < 4; t++) {
            int col = chunk * 128 + (cthalf * 4 + t) * 16 + (lane & 15);
            float bias = fc_b[col] + 1e-10f;
#pragma unroll
            for (int r = 0; r < 8; r++) {
                int row = r0 + rt * 16 + mrow0 + r;
                out[(size_t)row * EMB_ + col] = acc[t][r] + bias;
            }
        }
    }
}

// =====================================================================
// Kernel 3: in-place row L2 normalization
// =====================================================================
__global__ __launch_bounds__(256) void k3_norm(float* __restrict__ out)
{
    const int tid = threadIdx.x;
    float* p = out + (size_t)blockIdx.x * EMB_;
    float4 v = ((const float4*)p)[tid];
    float s = v.x * v.x + v.y * v.y + v.z * v.z + v.w * v.w;
    __shared__ float red[256];
    red[tid] = s;
    __syncthreads();
    for (int off = 128; off > 0; off >>= 1) {
        if (tid < off) red[tid] += red[tid + off];
        __syncthreads();
    }
    float inv = 1.0f / (sqrtf(red[0]) + 1e-8f);
    v.x *= inv; v.y *= inv; v.z *= inv; v.w *= inv;
    ((float4*)p)[tid] = v;
}

// =====================================================================
extern "C" void kernel_launch(void* const* d_in, const int* in_sizes, int n_in,
                              void* d_out, int out_size, void* d_ws, size_t ws_size,
                              hipStream_t stream)
{
    const int*   captions = (const int*)d_in[0];
    const float* hidden   = (const float*)d_in[1];
    const int*   tfeat    = (const int*)d_in[2];
    const float* fmask    = (const float*)d_in[3];
    const float* Wa       = (const float*)d_in[4];
    const float* Wc       = (const float*)d_in[5];
    const float* fc_w     = (const float*)d_in[6];
    const float* fc_b     = (const float*)d_in[7];
    float* out = (float*)d_out;

    // workspace layout (bf16 buffers)
    char* ws = (char*)d_ws;
    __bf16* oBuf  = (__bf16*)(ws);                      // 16 MB [8192,1024]
    __bf16* hidBf = (__bf16*)(ws + (16u << 20));        // 16 MB [8192,1024]
    __bf16* fcwBf = (__bf16*)(ws + (32u << 20));        //  4 MB [1024,2048]
    __bf16* eaBuf = (__bf16*)(ws + (36u << 20));        // 16 MB [8192,1024]
    __bf16* ecT   = (__bf16*)(ws + (52u << 20));        // 16 MB [32,1024,256]

    k0_cvt<<<dim3((B_ * LQ_ * HID_) / 1024), dim3(256), 0, stream>>>(hidden, hidBf,
                                                                     B_ * LQ_ * HID_);
    k0_cvt<<<dim3((EMB_ * 2 * HID_) / 1024), dim3(256), 0, stream>>>(fc_w, fcwBf,
                                                                     EMB_ * 2 * HID_);
    k0_gather<<<dim3(B_ * LKV_), dim3(256), 0, stream>>>(captions, tfeat, Wa, Wc,
                                                         eaBuf, ecT);
    k1_attn<<<dim3(B_ * (LQ_ / 64)), dim3(256), 0, stream>>>(hidBf, eaBuf, ecT,
                                                             fmask, oBuf);
    k2_fc<<<dim3((B_ * LQ_) / 64), dim3(256), 0, stream>>>(hidBf, oBuf, fcwBf, fc_b, out);
    k3_norm<<<dim3(B_ * LQ_), dim3(256), 0, stream>>>(out);
}
